// MultiHeadAttention_81174881894814
// MI455X (gfx1250) — compile-verified, matrix operand order checked by reading, not run on hardware
//
#include <hip/hip_runtime.h>
#include <hip/hip_bf16.h>

typedef __bf16 bf16_t;
typedef bf16_t v16bf __attribute__((ext_vector_type(16)));
typedef float  v8f   __attribute__((ext_vector_type(8)));
typedef unsigned short u16;
typedef unsigned int   u32;

// Problem constants (fixed by reference setup)
#define DMODEL 1024
#define NHEAD  16
#define DK     64
#define NB     2
#define SEQ    2048
#define MTOT   (NB * SEQ)   // 4096

#if __has_builtin(__builtin_amdgcn_global_load_async_to_lds_b128)
#define HAVE_ASYNC_LDS 1
#endif

// ---------------------------------------------------------------------------
// helpers
// ---------------------------------------------------------------------------
__device__ __forceinline__ u16 f2bf(float x) {
    bf16_t h = (bf16_t)x;                 // native v_cvt_pk_bf16_f32 on gfx1250
    return __builtin_bit_cast(u16, h);
}

__device__ __forceinline__ v8f wmma_bf16(v16bf a, v16bf b, v8f c) {
    // (neg_a, A, neg_b, B, c_mod, C, reuse_a, reuse_b)
    return __builtin_amdgcn_wmma_f32_16x16x32_bf16(false, a, false, b,
                                                   (short)0, c, false, false);
}

// 16B global->LDS copy: async (ASYNCcnt) when available, else via VGPRs.
#ifdef HAVE_ASYNC_LDS
typedef int v4i_t __attribute__((vector_size(16)));
typedef __attribute__((address_space(1))) v4i_t* gas_v4i;
typedef __attribute__((address_space(3))) v4i_t* las_v4i;
#endif

__device__ __forceinline__ void cp16_g2l(const u16* __restrict__ g, u16* l) {
#ifdef HAVE_ASYNC_LDS
    __builtin_amdgcn_global_load_async_to_lds_b128((gas_v4i)g, (las_v4i)l, 0, 0);
#else
    *(uint4*)l = *(const uint4*)g;
#endif
}

__device__ __forceinline__ void async_wait() {
#ifdef HAVE_ASYNC_LDS
    asm volatile("s_wait_asynccnt 0" ::: "memory");
#endif
}

// Load one 16x32 bf16 A/B fragment (A layout; B fed as N x K == B^T row-major).
__device__ __forceinline__ v16bf frag_ld(const u16* base, int row0, int stride, int kb) {
    const int lane = threadIdx.x & 31;
    const int half = lane >> 4;        // K-half select
    const int lr   = lane & 15;        // matrix row (A) / N index (B)
    const u16* p = base + (size_t)(row0 + lr) * stride + kb + (half << 3);
    union { uint4 u[2]; v16bf v; } f;
    f.u[0] = *(const uint4*)(p);        // K =  kb+half*8 .. +7
    f.u[1] = *(const uint4*)(p + 16);   // K =  kb+16+half*8 .. +7
    return f.v;
}

union V8 { v8f v; float f[8]; };

// ---------------------------------------------------------------------------
// Kernel 1: C[bf16, MxN] = A[f32, MxK] @ W[f32, KxN]      M=4096, K=N=1024
// 128(M) x 256(N) block tile, 8 waves in 2x4; each wave owns a 64x64 register
// tile: 16 WMMAs per 16 ds_load_b128 (2.0 WMMA per fragment pair).
// ---------------------------------------------------------------------------
__global__ __launch_bounds__(256)
void proj_gemm_kernel(const float* __restrict__ A, const float* __restrict__ W,
                      u16* __restrict__ C) {
    __shared__ __align__(16) u16 As[128 * 32];   // [m][k] bf16
    __shared__ __align__(16) u16 Bs[256 * 32];   // [n][k] bf16 (W transposed)

    const int t      = threadIdx.x;
    const int wv     = t >> 5;
    const int blockM = blockIdx.y * 128;
    const int blockN = blockIdx.x * 256;
    const int m0w    = (wv >> 2) * 64;    // 2 m-groups of 64 (4 tiles)
    const int n0w    = (wv & 3) * 64;     // 4 n-groups of 64 (4 tiles)

    V8 acc[4][4] = {};

    for (int k0 = 0; k0 < DMODEL; k0 += 32) {
        __syncthreads();
        {   // A tile: 128x32 f32 -> bf16, row-major
            const int row = t >> 1, c0 = (t & 1) * 16;
            const float* src = A + (size_t)(blockM + row) * DMODEL + k0 + c0;
            if (k0 + 32 < DMODEL) __builtin_prefetch(src + 32, 0, 1);
            u16 tmp[16] __attribute__((aligned(16)));
            #pragma unroll
            for (int i = 0; i < 16; i += 4) {
                float4 f = *(const float4*)(src + i);
                tmp[i+0] = f2bf(f.x); tmp[i+1] = f2bf(f.y);
                tmp[i+2] = f2bf(f.z); tmp[i+3] = f2bf(f.w);
            }
            uint4* dst = (uint4*)&As[row * 32 + c0];
            dst[0] = ((const uint4*)tmp)[0];
            dst[1] = ((const uint4*)tmp)[1];
        }
        {   // W tile: 32x256 f32 (row-major read) -> Bs[n][k] bf16 (transpose)
            const int k = t >> 3, g = t & 7;            // 32 cols per thread
            const float* src = W + (size_t)(k0 + k) * DMODEL + blockN + g * 32;
            if (k0 + 32 < DMODEL) __builtin_prefetch(src + 32 * DMODEL, 0, 1);
            #pragma unroll
            for (int i = 0; i < 32; i += 4) {
                float4 f = *(const float4*)(src + i);
                Bs[(g*32 + i + 0) * 32 + k] = f2bf(f.x);
                Bs[(g*32 + i + 1) * 32 + k] = f2bf(f.y);
                Bs[(g*32 + i + 2) * 32 + k] = f2bf(f.z);
                Bs[(g*32 + i + 3) * 32 + k] = f2bf(f.w);
            }
        }
        __syncthreads();

        v16bf af[4];
        #pragma unroll
        for (int mi = 0; mi < 4; ++mi) af[mi] = frag_ld(As, m0w + mi * 16, 32, 0);
        #pragma unroll
        for (int ni = 0; ni < 4; ++ni) {
            v16bf bfm = frag_ld(Bs, n0w + ni * 16, 32, 0);
            #pragma unroll
            for (int mi = 0; mi < 4; ++mi)
                acc[mi][ni].v = wmma_bf16(af[mi], bfm, acc[mi][ni].v);
        }
    }

    const int lane = t & 31, half = lane >> 4, lr = lane & 15;
    #pragma unroll
    for (int mi = 0; mi < 4; ++mi)
        #pragma unroll
        for (int ni = 0; ni < 4; ++ni) {
            const int col = blockN + n0w + ni * 16 + lr;
            #pragma unroll
            for (int r = 0; r < 8; ++r) {
                const int row = blockM + m0w + mi * 16 + r + 8 * half;
                C[(size_t)row * DMODEL + col] = f2bf(acc[mi][ni].f[r]);
            }
        }
}

// ---------------------------------------------------------------------------
// Kernel 2: flash attention per (batch, head, 64-row q tile).
// 4 waves; wave wv owns q rows [wv*16, wv*16+16). Online softmax is a pure
// half-wave (16 lane) shuffle reduction -> no cross-wave reduction needed.
// ---------------------------------------------------------------------------
__global__ __launch_bounds__(128)
void attn_kernel(const u16* __restrict__ Qp, const u16* __restrict__ Kp,
                 const u16* __restrict__ Vp, u16* __restrict__ Ob) {
    __shared__ __align__(16) u16 Ks[64 * 64];      // [key][dk]   (= B N x K for QK^T)
    __shared__ __align__(16) u16 Vs[64 * 64];      // [dv][key]   (= B N x K for PV)
    __shared__ __align__(16) u16 Ps[4 * 16 * 64];  // per-wave P scratch

    const int t    = threadIdx.x;
    const int wv   = t >> 5;
    const int lane = t & 31, half = lane >> 4, lr = lane & 15;
    const int h    = blockIdx.y;
    const int b    = blockIdx.z;
    const int q0   = blockIdx.x * 64;
    const size_t rowBase = (size_t)b * SEQ;

    // Q fragments: 16 rows x 64 dk, straight from global (bf16, row stride 1024)
    v16bf qf[2];
    {
        const u16* qb = Qp + (rowBase + q0 + wv * 16) * DMODEL + h * DK;
        qf[0] = frag_ld(qb, 0, DMODEL, 0);
        qf[1] = frag_ld(qb, 0, DMODEL, 32);
    }

    V8 o[4] = {};
    float m[8], l[8];
    #pragma unroll
    for (int r = 0; r < 8; ++r) { m[r] = -1e30f; l[r] = 0.0f; }

    u16* Pw = Ps + wv * (16 * 64);

    for (int key0 = 0; key0 < SEQ; key0 += 64) {
        __syncthreads();
        {   // K tile: [key][dk] async copy, 2 threads per row, 64B each
            const int key = t >> 1, part = t & 1;
            const u16* src = Kp + (rowBase + key0 + key) * DMODEL + h * DK + part * 32;
            u16* dst = &Ks[key * 64 + part * 32];
            #pragma unroll
            for (int i = 0; i < 4; ++i) cp16_g2l(src + i * 8, dst + i * 8);
        }
        {   // V tile: transpose into Vs[dv][key] (must go through VGPRs)
            const int key = t & 63, dh = t >> 6;
            const u16* src = Vp + (rowBase + key0 + key) * DMODEL + h * DK + dh * 32;
            u16 tmp[32] __attribute__((aligned(16)));
            #pragma unroll
            for (int i = 0; i < 4; ++i) ((uint4*)tmp)[i] = ((const uint4*)src)[i];
            #pragma unroll
            for (int j = 0; j < 32; ++j) Vs[(dh * 32 + j) * 64 + key] = tmp[j];
        }
        async_wait();
        __syncthreads();

        // S = Q @ K^T : 8 WMMAs
        V8 s[4] = {};
        #pragma unroll
        for (int kc = 0; kc < 2; ++kc)
            #pragma unroll
            for (int nt = 0; nt < 4; ++nt) {
                v16bf kf = frag_ld(Ks, nt * 16, 64, kc * 32);
                s[nt].v = wmma_bf16(qf[kc], kf, s[nt].v);
            }

        #pragma unroll
        for (int nt = 0; nt < 4; ++nt)
            #pragma unroll
            for (int r = 0; r < 8; ++r) s[nt].f[r] *= 0.125f;  // 1/sqrt(DK)

        // online softmax per row (rows live in a 16-lane half)
        #pragma unroll
        for (int r = 0; r < 8; ++r) {
            float mx = fmaxf(fmaxf(s[0].f[r], s[1].f[r]),
                             fmaxf(s[2].f[r], s[3].f[r]));
            #pragma unroll
            for (int off = 1; off < 16; off <<= 1)
                mx = fmaxf(mx, __shfl_xor(mx, off, 32));
            const float mnew = fmaxf(m[r], mx);
            const float a    = __expf(m[r] - mnew);
            m[r] = mnew;
            float rs = 0.0f;
            #pragma unroll
            for (int nt = 0; nt < 4; ++nt) {
                const float p = __expf(s[nt].f[r] - mnew);
                s[nt].f[r] = p;
                rs += p;
            }
            #pragma unroll
            for (int off = 1; off < 16; off <<= 1)
                rs += __shfl_xor(rs, off, 32);
            l[r] = l[r] * a + rs;
            #pragma unroll
            for (int nt = 0; nt < 4; ++nt) o[nt].f[r] *= a;
        }

        // accumulator layout -> A layout via wave-private LDS strip
        #pragma unroll
        for (int nt = 0; nt < 4; ++nt)
            #pragma unroll
            for (int r = 0; r < 8; ++r)
                Pw[(r + 8 * half) * 64 + nt * 16 + lr] = f2bf(s[nt].f[r]);

        // O += P @ V : 8 WMMAs
        #pragma unroll
        for (int kc = 0; kc < 2; ++kc) {
            v16bf pf = frag_ld(Pw, 0, 64, kc * 32);
            #pragma unroll
            for (int nt = 0; nt < 4; ++nt) {
                v16bf vf = frag_ld(Vs, nt * 16, 64, kc * 32);
                o[nt].v = wmma_bf16(pf, vf, o[nt].v);
            }
        }
    }

    float inv[8];
    #pragma unroll
    for (int r = 0; r < 8; ++r) inv[r] = 1.0f / l[r];
    #pragma unroll
    for (int nt = 0; nt < 4; ++nt)
        #pragma unroll
        for (int r = 0; r < 8; ++r) {
            const size_t row = rowBase + q0 + wv * 16 + r + 8 * half;
            Ob[row * DMODEL + h * DK + nt * 16 + lr] = f2bf(o[nt].f[r] * inv[r]);
        }
}

// ---------------------------------------------------------------------------
// Kernel 3: out[f32] = attn[bf16] @ WO[f32->bf16] + bO
// Same 128x256 / 64x64-per-wave shape as Kernel 1.
// ---------------------------------------------------------------------------
__global__ __launch_bounds__(256)
void out_gemm_kernel(const u16* __restrict__ A, const float* __restrict__ W,
                     const float* __restrict__ bias, float* __restrict__ C) {
    __shared__ __align__(16) u16 As[128 * 32];
    __shared__ __align__(16) u16 Bs[256 * 32];

    const int t      = threadIdx.x;
    const int wv     = t >> 5;
    const int blockM = blockIdx.y * 128;
    const int blockN = blockIdx.x * 256;
    const int m0w    = (wv >> 2) * 64;
    const int n0w    = (wv & 3) * 64;

    V8 acc[4][4] = {};

    for (int k0 = 0; k0 < DMODEL; k0 += 32) {
        __syncthreads();
        {   // A tile: already bf16, async copy
            const int row = t >> 1, part = t & 1;
            const u16* src = A + (size_t)(blockM + row) * DMODEL + k0 + part * 16;
            u16* dst = &As[row * 32 + part * 16];
            cp16_g2l(src, dst);
            cp16_g2l(src + 8, dst + 8);
            if (k0 + 32 < DMODEL) __builtin_prefetch(src + 32, 0, 1);
        }
        {   // WO tile: 32x256 f32 -> Bs[n][k] bf16
            const int k = t >> 3, g = t & 7;
            const float* src = W + (size_t)(k0 + k) * DMODEL + blockN + g * 32;
            if (k0 + 32 < DMODEL) __builtin_prefetch(src + 32 * DMODEL, 0, 1);
            #pragma unroll
            for (int i = 0; i < 32; i += 4) {
                float4 f = *(const float4*)(src + i);
                Bs[(g*32 + i + 0) * 32 + k] = f2bf(f.x);
                Bs[(g*32 + i + 1) * 32 + k] = f2bf(f.y);
                Bs[(g*32 + i + 2) * 32 + k] = f2bf(f.z);
                Bs[(g*32 + i + 3) * 32 + k] = f2bf(f.w);
            }
        }
        async_wait();
        __syncthreads();

        v16bf af[4];
        #pragma unroll
        for (int mi = 0; mi < 4; ++mi) af[mi] = frag_ld(As, m0w + mi * 16, 32, 0);
        #pragma unroll
        for (int ni = 0; ni < 4; ++ni) {
            v16bf bfm = frag_ld(Bs, n0w + ni * 16, 32, 0);
            #pragma unroll
            for (int mi = 0; mi < 4; ++mi)
                acc[mi][ni].v = wmma_bf16(af[mi], bfm, acc[mi][ni].v);
        }
    }

    const int lane = t & 31, half = lane >> 4, lr = lane & 15;
    #pragma unroll
    for (int mi = 0; mi < 4; ++mi)
        #pragma unroll
        for (int ni = 0; ni < 4; ++ni) {
            const int col = blockN + n0w + ni * 16 + lr;
            const float bv = bias[col];
            #pragma unroll
            for (int r = 0; r < 8; ++r) {
                const int row = blockM + m0w + mi * 16 + r + 8 * half;
                C[(size_t)row * DMODEL + col] = acc[mi][ni].f[r] + bv;
            }
        }
}

// ---------------------------------------------------------------------------
// host launcher
// ---------------------------------------------------------------------------
extern "C" void kernel_launch(void* const* d_in, const int* in_sizes, int n_in,
                              void* d_out, int out_size, void* d_ws, size_t ws_size,
                              hipStream_t stream) {
    const float* Q  = (const float*)d_in[0];
    const float* K  = (const float*)d_in[1];
    const float* V  = (const float*)d_in[2];
    const float* WQ = (const float*)d_in[3];
    const float* WK = (const float*)d_in[4];
    const float* WV = (const float*)d_in[5];
    const float* WO = (const float*)d_in[6];
    const float* bO = (const float*)d_in[7];

    // workspace: 4 x (4096*1024) bf16 intermediates = 32 MB
    u16* qp = (u16*)d_ws;
    u16* kp = qp + (size_t)MTOT * DMODEL;
    u16* vp = kp + (size_t)MTOT * DMODEL;
    u16* ab = vp + (size_t)MTOT * DMODEL;

    const dim3 gemm_grid(DMODEL / 256, MTOT / 128);   // (4, 32)

    proj_gemm_kernel<<<gemm_grid, 256, 0, stream>>>(Q, WQ, qp);
    proj_gemm_kernel<<<gemm_grid, 256, 0, stream>>>(K, WK, kp);
    proj_gemm_kernel<<<gemm_grid, 256, 0, stream>>>(V, WV, vp);

    attn_kernel<<<dim3(SEQ / 64, NHEAD, NB), 128, 0, stream>>>(qp, kp, vp, ab);

    out_gemm_kernel<<<gemm_grid, 256, 0, stream>>>(ab, WO, bO, (float*)d_out);
}
